// Malis_76012331204897
// MI455X (gfx1250) — 8-bit matrix operand order repaired and checked by reading, compiled, not run on hardware
//
#include <hip/hip_runtime.h>
#include <cstdint>

typedef __attribute__((ext_vector_type(8))) int v8i;

#define MALIS_H 128
#define MALIS_W 128
#define MALIS_L 64

__global__ void malis_init_kernel(const int* __restrict__ labels,
                                  unsigned short* __restrict__ counts,
                                  float* __restrict__ out, int total_nodes) {
    int t = blockIdx.x * blockDim.x + threadIdx.x;
    if (t == 0) out[0] = 0.0f;
    if (t < total_nodes) {
        int lab = labels[t] & (MALIS_L - 1);
        unsigned short* c = counts + (size_t)t * MALIS_L;
        #pragma unroll 8
        for (int l = 0; l < MALIS_L; ++l) c[l] = (l == lab) ? 1 : 0;
    }
}

__global__ void malis_sort_kernel(const float* __restrict__ aff,
                                  uint64_t* __restrict__ sorted, int E) {
    extern __shared__ uint64_t smem[];
    const int b = blockIdx.x;
    const float* a = aff + (size_t)b * E;

    for (int i = threadIdx.x; i < E; i += blockDim.x) {
        uint32_t m = __float_as_uint(a[i]);
        uint32_t asc = (m & 0x80000000u) ? ~m : (m ^ 0x80000000u);
        uint32_t key = ~asc;
        smem[i] = ((uint64_t)key << 32) | (uint32_t)i;
    }
    __syncthreads();

    for (int k = 2; k <= E; k <<= 1) {
        for (int j = k >> 1; j > 0; j >>= 1) {
            for (int t = threadIdx.x; t < E; t += blockDim.x) {
                int ixj = t ^ j;
                if (ixj > t) {
                    uint64_t x = smem[t], y = smem[ixj];
                    bool up = ((t & k) == 0);
                    if ((x > y) == up) { smem[t] = y; smem[ixj] = x; }
                }
            }
            __syncthreads();
        }
    }

    uint64_t* o = sorted + (size_t)b * E;
    for (int i = threadIdx.x; i < E; i += blockDim.x) o[i] = smem[i];
}

__global__ void malis_scan_kernel(const float* __restrict__ aff,
                                  const uint64_t* __restrict__ sorted,
                                  unsigned short* __restrict__ counts,
                                  float* __restrict__ out) {
    const int H = MALIS_H, W = MALIS_W, L = MALIS_L;
    const int N = H * W, E = 2 * N;
    const int b = blockIdx.x;
    const int lane = threadIdx.x;

    extern __shared__ int lds[];
    int* parent = lds;
    int* sz     = lds + N;
    unsigned char* stage = (unsigned char*)(lds + 2 * N);

    const float*    A   = aff    + (size_t)b * E;
    const uint64_t* ord = sorted + (size_t)b * E;
    unsigned short* cnt = counts + (size_t)b * N * L;

    for (int i = lane; i < N; i += 32) { parent[i] = i; sz[i] = 1; }
    {
        int l0 = 2 * lane, l1 = l0 + 1;
        stage[4 * 64 + l0] = (l0 == 0) ? 0 : 1;
        stage[4 * 64 + l1] = 1;
        stage[5 * 64 + l0] = 0;
        stage[5 * 64 + l1] = 0;
    }
    __syncthreads();

    const int m    = lane & 15;
    const int half = lane >> 4;
    const unsigned arow_off = (m == 0) ? 0u : (m == 1) ? 64u : (m == 2) ? 4u * 64u : 5u * 64u;
    const unsigned bcol_off = (m == 0) ? 2u * 64u : (m == 1) ? 3u * 64u : (m == 2) ? 4u * 64u : 5u * 64u;

    double acc = 0.0;
    for (int e = 0; e < E; ++e) {
        __builtin_prefetch(&ord[(e + 16 < E) ? e + 16 : e], 0, 0);
        uint64_t rec = ord[e];
        int idx = (int)(uint32_t)rec;

        int eu, ev;
        if (idx < N) {
            int i = idx / W;
            bool ok = (i < H - 1);
            eu = ok ? idx : 0;
            ev = ok ? idx + W : 0;
        } else {
            int p = idx - N;
            int j = p % W;
            bool ok = (j < W - 1);
            eu = ok ? p : 0;
            ev = ok ? p + 1 : 0;
        }

        int ru = eu; while (parent[ru] != ru) ru = parent[ru];
        int rv = ev; while (parent[rv] != rv) rv = parent[rv];

        if (ru != rv) {
            float a = A[idx];
            int l0 = 2 * lane, l1 = l0 + 1;
            int cu0 = (l0 == 0) ? 0 : (int)cnt[ru * L + l0];
            int cu1 = (int)cnt[ru * L + l1];
            int cv0 = (l0 == 0) ? 0 : (int)cnt[rv * L + l0];
            int cv1 = (int)cnt[rv * L + l1];
            stage[0 * 64 + l0] = (unsigned char)(cu0 & 255);
            stage[0 * 64 + l1] = (unsigned char)(cu1 & 255);
            stage[1 * 64 + l0] = (unsigned char)(cu0 >> 8);
            stage[1 * 64 + l1] = (unsigned char)(cu1 >> 8);
            stage[2 * 64 + l0] = (unsigned char)(cv0 & 255);
            stage[2 * 64 + l1] = (unsigned char)(cv1 & 255);
            stage[3 * 64 + l0] = (unsigned char)(cv0 >> 8);
            stage[3 * 64 + l1] = (unsigned char)(cv1 >> 8);
            __syncthreads();

            const unsigned int* arow = (const unsigned int*)(stage + arow_off);
            const unsigned int* bcol = (const unsigned int*)(stage + bcol_off);
            v8i av, bv, cz = {0, 0, 0, 0, 0, 0, 0, 0};
            #pragma unroll
            for (int g = 0; g < 4; ++g) {
                int base = (g / 2) * 32 + half * 16 + (g % 2) * 8;
                av[2 * g]     = (int)arow[base / 4];
                av[2 * g + 1] = (int)arow[base / 4 + 1];
            }
            #pragma unroll
            for (int j2 = 0; j2 < 8; ++j2) {
                int kbase = (j2 / 4) * 32 + half * 16 + (j2 % 4) * 4;
                bv[j2] = (int)bcol[kbase / 4];
            }

            v8i d = __builtin_amdgcn_wmma_i32_16x16x64_iu8(
                false, av, false, bv, cz, false, false);

            long long S00 = __shfl(d[0], 0, 32);
            long long S01 = __shfl(d[0], 1, 32);
            long long S02 = __shfl(d[0], 2, 32);
            long long S10 = __shfl(d[1], 0, 32);
            long long S11 = __shfl(d[1], 1, 32);
            long long S12 = __shfl(d[1], 2, 32);
            long long S20 = __shfl(d[2], 0, 32);
            long long S21 = __shfl(d[2], 1, 32);

            long long same = S00 + 256LL * (S01 + S10) + 65536LL * S11;
            long long tu   = S02 + 256LL * S12;
            long long tv   = S20 + 256LL * S21;
            long long npos = same;
            long long nneg = tu * tv - same;

            acc += (double)npos * ((double)a - 1.0) + (double)nneg * (double)a;

            int big   = (sz[ru] >= sz[rv]) ? ru : rv;
            int small = ru + rv - big;
            cnt[big * L + l0] = (unsigned short)(cnt[big * L + l0] + cnt[small * L + l0]);
            cnt[big * L + l1] = (unsigned short)(cnt[big * L + l1] + cnt[small * L + l1]);
            if (lane == 0) { parent[small] = big; sz[big] += sz[small]; }
            __syncthreads();
        }
    }

    if (lane == 0) atomicAdd(out, (float)(-0.5 * acc));
}

extern "C" void kernel_launch(void* const* d_in, const int* in_sizes, int n_in,
                              void* d_out, int out_size, void* d_ws, size_t ws_size,
                              hipStream_t stream) {
    const float* aff    = (const float*)d_in[0];
    const int*   labels = (const int*)d_in[1];
    float*       out    = (float*)d_out;

    const int H = MALIS_H, W = MALIS_W;
    const int N = H * W;
    const int E = 2 * N;
    const int B = (n_in > 1 && in_sizes[1] > 0) ? (in_sizes[1] / N) : 2;

    uint64_t*       sorted = (uint64_t*)d_ws;
    unsigned short* counts = (unsigned short*)((char*)d_ws + (size_t)B * E * sizeof(uint64_t));

    const size_t sortLds = (size_t)E * sizeof(uint64_t);
    const size_t scanLds = (size_t)2 * N * sizeof(int) + 6 * 64;
    (void)hipFuncSetAttribute((const void*)malis_sort_kernel,
                              hipFuncAttributeMaxDynamicSharedMemorySize, (int)sortLds);
    (void)hipFuncSetAttribute((const void*)malis_scan_kernel,
                              hipFuncAttributeMaxDynamicSharedMemorySize, (int)scanLds);

    int totalNodes = B * N;
    malis_init_kernel<<<(totalNodes + 255) / 256, 256, 0, stream>>>(labels, counts, out, totalNodes);
    malis_sort_kernel<<<B, 1024, sortLds, stream>>>(aff, sorted, E);
    malis_scan_kernel<<<B, 32, scanLds, stream>>>(aff, sorted, counts, out);
}
